// InfiniteMambaAudioProcessor_84722524881487
// MI455X (gfx1250) — matrix operand order repaired and checked by reading, not compiled, not run on hardware
//
#include <hip/hip_runtime.h>

// ---- fixed problem shapes ----
#define B_     4
#define C_     4
#define T_     4096
#define H_     256
#define INNER_ 512
#define S_     64
#define L_     3
#define V_     1024
#define M_     (B_*T_)   // 16384 rows (b,t)

typedef __attribute__((ext_vector_type(16))) __bf16 v16bf;
typedef __attribute__((ext_vector_type(8)))  float  v8f;
typedef __attribute__((ext_vector_type(4)))  unsigned u32x4;
typedef __attribute__((ext_vector_type(4)))  int      i32x4;
typedef unsigned short u16b;

__device__ __forceinline__ u16b f2bf(float f){
  unsigned u = __float_as_uint(f);
  unsigned r = u + 0x7FFFu + ((u >> 16) & 1u);
  return (u16b)(r >> 16);
}
__device__ __forceinline__ float bf2f(u16b h){ return __uint_as_float(((unsigned)h) << 16); }
__device__ __forceinline__ float sigm(float x){ return 1.f/(1.f + __expf(-x)); }

union Frag { v16bf v; u32x4 u[2]; };

// ---- async global->LDS copy (gfx1250), guarded with fallback ----
#if __has_builtin(__builtin_amdgcn_global_load_async_to_lds_b128)
#define HAVE_ASYNC_LDS 1
__device__ __forceinline__ void cp_g2l_16(const u16b* g, u16b* l){
  u16b* gg = const_cast<u16b*>(g);
  i32x4* gv = reinterpret_cast<i32x4*>(gg);
  i32x4* lv = reinterpret_cast<i32x4*>(l);
  __builtin_amdgcn_global_load_async_to_lds_b128(
      (__attribute__((address_space(1))) i32x4*)gv,
      (__attribute__((address_space(3))) i32x4*)lv, 0, 0);
}
__device__ __forceinline__ void cp_wait(){
#if __has_builtin(__builtin_amdgcn_s_wait_asynccnt)
  __builtin_amdgcn_s_wait_asynccnt(0);
#else
  asm volatile("s_wait_asynccnt 0" ::: "memory");
#endif
  __syncthreads();
}
#else
#define HAVE_ASYNC_LDS 0
__device__ __forceinline__ void cp_g2l_16(const u16b* g, u16b* l){
  *(u32x4*)l = *(const u32x4*)g;
}
__device__ __forceinline__ void cp_wait(){ __syncthreads(); }
#endif

// ---------------- generic bf16 WMMA GEMM: D = A(MxK) * W(NxK)^T ----------------
// block = 8 waves x 32 rows = 256 rows, all waves share one 64-col W tile in LDS
enum { EPI_BF=0, EPI_GATE=1, EPI_RESID=2, EPI_RELU=3, EPI_HEAD=4 };

template<int EPI>
__global__ __launch_bounds__(256) void gemm_wmma(
    const u16b* __restrict__ A0, const u16b* __restrict__ A1,
    const u16b* __restrict__ W, int N, int K, int lda,
    u16b* __restrict__ outB, float* __restrict__ outF,
    const float* __restrict__ bias, const float* __restrict__ Dvec,
    const u16b* __restrict__ xg, const u16b* __restrict__ xcg,
    const u16b* __restrict__ resid)
{
  extern __shared__ u16b sW[];              // 64 x (K+8) bf16, dynamic
  const int ntN = N >> 6;
  int bm = blockIdx.x / ntN;                // 64 row-blocks of 256 rows
  int tn = blockIdx.x % ntN;
  int tm = (bm << 3) + (threadIdx.x >> 5);  // wave's 32-row tile
  int lane = threadIdx.x & 31;
  int lh   = lane >> 4, lm = lane & 15;
  const int ldw = K + 8;                    // padded LDS stride (bank-spread)

  // ---- cooperative stage of W tile [tn*64, tn*64+64) x K into LDS ----
  {
    const int chunks = (64 * K) >> 3;       // 16B chunks
    const int perrow = K >> 3;
    for (int c = threadIdx.x; c < chunks; c += 256){
      int col = c / perrow, kc = (c % perrow) << 3;
      cp_g2l_16(W + (size_t)(tn*64 + col)*K + kc, sW + col*ldw + kc);
    }
    cp_wait();
  }

  v8f acc[2][4];
  #pragma unroll
  for (int a=0;a<2;a++)
    #pragma unroll
    for (int b=0;b<4;b++)
      #pragma unroll
      for (int r=0;r<8;r++) acc[a][b][r]=0.f;

  for (int kk=0; kk<K; kk+=32){
    const u16b* Ab = A0; int kl = kk;
    if (EPI==EPI_RELU && kk >= 256){ Ab = A1; kl = kk - 256; }   // concat [feat|pat]

    Frag af[2];
    #pragma unroll
    for (int mf=0; mf<2; mf++){
      int row = (tm<<5) + (mf<<4) + lm;
      const u16b* p = Ab + (size_t)row*lda + kl + (lh<<3);
      af[mf].u[0] = *(const u32x4*)p;
      af[mf].u[1] = *(const u32x4*)(p + 16);
    }
    Frag bfr[4];
    #pragma unroll
    for (int nf=0; nf<4; nf++){
      const u16b* q = sW + (size_t)((nf<<4)+lm)*ldw + kk + (lh<<3);
      bfr[nf].u[0] = *(const u32x4*)q;
      bfr[nf].u[1] = *(const u32x4*)(q + 16);
    }
    if (kk + 32 < K)
      __builtin_prefetch(Ab + (size_t)((tm<<5)+lm)*lda + kl + 32, 0, 1);

    #pragma unroll
    for (int mf=0; mf<2; mf++)
      #pragma unroll
      for (int nf=0; nf<4; nf++)
        acc[mf][nf] = __builtin_amdgcn_wmma_f32_16x16x32_bf16(
            false, af[mf].v, false, bfr[nf].v, (short)0, acc[mf][nf], false, false);
  }

  #pragma unroll
  for (int mf=0; mf<2; mf++){
    #pragma unroll
    for (int nf=0; nf<4; nf++){
      #pragma unroll
      for (int r=0; r<8; r++){
        int m = (tm<<5) + (mf<<4) + (lh<<3) + r;
        int n = (tn<<6) + (nf<<4) + lm;
        float v = acc[mf][nf][r];
        if (EPI==EPI_BF){
          outB[(size_t)m*N + n] = f2bf(v);
        } else if (EPI==EPI_GATE){
          float xcv = bf2f(xcg[(size_t)m*INNER_ + n]);
          v += Dvec[n] * xcv;
          float g = bf2f(xg[(size_t)m*(2*INNER_) + n]);   // xg base = XP + INNER_
          v *= g * sigm(g);                               // * silu(xg)
          outB[(size_t)m*INNER_ + n] = f2bf(v);
        } else if (EPI==EPI_RESID){
          outF[(size_t)m*H_ + n] = v + bf2f(resid[(size_t)m*H_ + n]);
        } else if (EPI==EPI_RELU){
          float t = v + bias[n];
          outF[(size_t)m*H_ + n] = t > 0.f ? t : 0.f;
        } else { // EPI_HEAD: scatter to (b,c,t,v)
          int b = m >> 12, t = m & (T_-1);
          int c = n >> 10, vv = n & (V_-1);
          outF[ (((size_t)(b*C_ + c))*T_ + t)*V_ + vv ] = v + bias[n];
        }
      }
    }
  }
}

// ---------------- dilated pattern conv as predicated WMMA GEMM ----------------
// block = 8 waves x 32 rows, one dilation; per-tap 64x256 W tile staged in LDS
__global__ __launch_bounds__(256) void pat_gemm(
    const u16b* __restrict__ EMB, const u16b* __restrict__ WPAT,
    const float* __restrict__ pat_b, u16b* __restrict__ PAT)
{
  extern __shared__ u16b sW[];              // 64 x (256+8)
  const int dils[4] = {1,3,6,12};
  int di = blockIdx.y;
  int d  = dils[di];
  int tm = (blockIdx.x << 3) + (threadIdx.x >> 5);
  int lane = threadIdx.x & 31, lh = lane >> 4, lm = lane & 15;
  const int ldw = H_ + 8;

  v8f acc[2][4];
  #pragma unroll
  for (int a=0;a<2;a++)
    #pragma unroll
    for (int b=0;b<4;b++)
      #pragma unroll
      for (int r=0;r<8;r++) acc[a][b][r]=0.f;

  for (int k=0; k<3; k++){
    int shift = (k-1)*d;
    const u16b* Wk = WPAT + (size_t)(di*3 + k)*64*H_;
    // stage this tap's 64x256 weight tile
    if (k) __syncthreads();                 // previous tap's reads done
    for (int c = threadIdx.x; c < (64*H_)>>3; c += 256){
      int col = c / (H_>>3), kc = (c % (H_>>3)) << 3;
      cp_g2l_16(Wk + (size_t)col*H_ + kc, sW + col*ldw + kc);
    }
    cp_wait();

    for (int kk=0; kk<H_; kk+=32){
      Frag af[2];
      #pragma unroll
      for (int mf=0; mf<2; mf++){
        int row = (tm<<5) + (mf<<4) + lm;
        int t   = row & (T_-1);
        int ts  = t + shift;
        u32x4 z = {0u,0u,0u,0u};
        af[mf].u[0] = z; af[mf].u[1] = z;
        if ((unsigned)ts < (unsigned)T_){
          const u16b* p = EMB + ((long)row + shift)*H_ + kk + (lh<<3);
          af[mf].u[0] = *(const u32x4*)p;
          af[mf].u[1] = *(const u32x4*)(p + 16);
        }
      }
      Frag bfr[4];
      #pragma unroll
      for (int nf=0; nf<4; nf++){
        const u16b* q = sW + (size_t)((nf<<4)+lm)*ldw + kk + (lh<<3);
        bfr[nf].u[0] = *(const u32x4*)q;
        bfr[nf].u[1] = *(const u32x4*)(q + 16);
      }
      #pragma unroll
      for (int mf=0; mf<2; mf++)
        #pragma unroll
        for (int nf=0; nf<4; nf++)
          acc[mf][nf] = __builtin_amdgcn_wmma_f32_16x16x32_bf16(
              false, af[mf].v, false, bfr[nf].v, (short)0, acc[mf][nf], false, false);
    }
  }
  #pragma unroll
  for (int mf=0; mf<2; mf++){
    #pragma unroll
    for (int nf=0; nf<4; nf++){
      #pragma unroll
      for (int r=0; r<8; r++){
        int m = (tm<<5) + (mf<<4) + (lh<<3) + r;
        int n = (nf<<4) + lm;                       // 0..63
        PAT[(size_t)m*H_ + (di<<6) + n] = f2bf(acc[mf][nf][r] + pat_b[(di<<6) + n]);
      }
    }
  }
}

// ---------------- small kernels ----------------
__global__ void cvt_f2bf(const float* __restrict__ s, u16b* __restrict__ d, int n){
  int i = blockIdx.x*256 + threadIdx.x;
  if (i < n) d[i] = f2bf(s[i]);
}

// pat_w (4,64,256,3) -> WPAT[(g*3+k)][o][h] bf16
__global__ void repack_pat(const float* __restrict__ pw, u16b* __restrict__ dst){
  int i = blockIdx.x*256 + threadIdx.x;
  if (i >= 4*64*256*3) return;
  int k = i % 3; int h = (i/3) % 256; int o = (i/(3*256)) % 64; int g = i/(3*256*64);
  dst[ ((size_t)(g*3 + k)*64 + o)*256 + h ] = f2bf(pw[i]);
}

__global__ __launch_bounds__(256) void embed_kernel(
    const int* __restrict__ tok, const float* __restrict__ ctx,
    const float* __restrict__ temb, const float* __restrict__ pemb,
    const int* __restrict__ poff, u16b* __restrict__ EMB, u16b* __restrict__ FEAT)
{
  int m = blockIdx.x, h = threadIdx.x;
  int b = m >> 12, t = m & (T_-1);
  float s = 0.f;
  #pragma unroll
  for (int c=0; c<C_; c++){
    int tk = tok[ (size_t)(b*C_ + c)*T_ + t ];
    s += temb[ ((size_t)c*V_ + tk)*H_ + h ];
  }
  s = s*0.25f + pemb[ ((size_t)(poff[0] + t))*H_ + h ] + ctx[(size_t)b*H_ + h];
  u16b bv = f2bf(s);
  EMB[(size_t)m*H_ + h]  = bv;
  FEAT[(size_t)m*H_ + h] = bv;
}

// depthwise causal conv k=4 over xi (=XP cols 0..511), + silu
__global__ void dwconv(const u16b* __restrict__ XP, const float* __restrict__ cw,
                       const float* __restrict__ cb, u16b* __restrict__ XC){
  int i = blockIdx.x*256 + threadIdx.x;      // M_*INNER_
  int m = i >> 9, ch = i & 511;
  int t = m & (T_-1);
  float acc = cb[ch];
  #pragma unroll
  for (int k=0; k<4; k++){
    int tt = t - 3 + k;
    if (tt >= 0) acc += bf2f(XP[(size_t)(m-3+k)*1024 + ch]) * cw[ch*4 + k];
  }
  XC[(size_t)m*512 + ch] = f2bf(acc * sigm(acc));
}

// chunked parallel scan: h = 0.95 h + 0.05 u, 256 recurrences, 32 chunks of 128
#define NC_ 32
#define CL_ 128
__global__ void scan_p1(const u16b* __restrict__ U, float* __restrict__ carry){
  int tid = blockIdx.x*256 + threadIdx.x;            // B*NC*S = 8192
  int s = tid & 63, chunk = (tid>>6) & 31, b = tid >> 11;
  int m0 = b*T_ + chunk*CL_;
  float h = 0.f;
  for (int j=0; j<CL_; j++) h = 0.95f*h + 0.05f*bf2f(U[(size_t)(m0+j)*64 + s]);
  carry[tid] = h;
}
__global__ void scan_p2(const float* __restrict__ carry, float* __restrict__ hin){
  int tid = threadIdx.x;                              // 256 = b*64+s
  int s = tid & 63, b = tid >> 6;
  float d128 = __powf(0.95f, (float)CL_);
  float run = 0.f;
  for (int c=0; c<NC_; c++){
    int idx = (b*NC_ + c)*64 + s;
    hin[idx] = run;
    run = run*d128 + carry[idx];
  }
}
__global__ void scan_p3(const u16b* __restrict__ U, const float* __restrict__ hin,
                        u16b* __restrict__ HS){
  int tid = blockIdx.x*256 + threadIdx.x;
  int s = tid & 63, chunk = (tid>>6) & 31, b = tid >> 11;
  int m0 = b*T_ + chunk*CL_;
  float h = hin[tid];
  for (int j=0; j<CL_; j++){
    h = 0.95f*h + 0.05f*bf2f(U[(size_t)(m0+j)*64 + s]);
    HS[(size_t)(m0+j)*64 + s] = f2bf(h);
  }
}

__global__ __launch_bounds__(256) void ln_kernel(const float* __restrict__ src,
    u16b* __restrict__ dst, const float* __restrict__ g, const float* __restrict__ bb){
  __shared__ float red[256];
  int m = blockIdx.x, h = threadIdx.x;
  float x = src[(size_t)m*256 + h];
  red[h] = x; __syncthreads();
  for (int s=128; s>0; s>>=1){ if (h<s) red[h]+=red[h+s]; __syncthreads(); }
  float mean = red[0]*(1.f/256.f);
  __syncthreads();
  float dd = x - mean; red[h] = dd*dd; __syncthreads();
  for (int s=128; s>0; s>>=1){ if (h<s) red[h]+=red[h+s]; __syncthreads(); }
  float var = red[0]*(1.f/256.f);
  dst[(size_t)m*256 + h] = f2bf(dd * rsqrtf(var + 1e-5f) * g[h] + bb[h]);
}

// ---------------- host launcher ----------------
extern "C" void kernel_launch(void* const* d_in, const int* in_sizes, int n_in,
                              void* d_out, int out_size, void* d_ws, size_t ws_size,
                              hipStream_t stream)
{
  const int*   tok    = (const int*)  d_in[0];
  const float* ctx    = (const float*)d_in[1];
  const float* temb   = (const float*)d_in[2];
  const float* pemb   = (const float*)d_in[3];
  const float* in_w   = (const float*)d_in[4];
  const float* conv_w = (const float*)d_in[5];
  const float* conv_b = (const float*)d_in[6];
  const float* Dp     = (const float*)d_in[7];
  const float* Bp_w   = (const float*)d_in[8];
  const float* Cp_w   = (const float*)d_in[9];
  const float* out_w  = (const float*)d_in[10];
  const float* ln_g   = (const float*)d_in[11];
  const float* ln_b   = (const float*)d_in[12];
  const float* pat_w  = (const float*)d_in[13];
  const float* pat_b  = (const float*)d_in[14];
  const float* fus_w  = (const float*)d_in[15];
  const float* fus_b  = (const float*)d_in[16];
  const float* fus_g  = (const float*)d_in[17];
  const float* fus_bb = (const float*)d_in[18];
  const float* head_w = (const float*)d_in[19];
  const float* head_b = (const float*)d_in[20];
  const int*   poff   = (const int*)  d_in[21];
  float* out = (float*)d_out;

  char* ws = (char*)d_ws;
  size_t off = 0;
  auto alloc = [&](size_t bytes)->char*{
    char* p = ws + off; off += (bytes + 255) & ~(size_t)255; return p;
  };
  u16b*  EMB   = (u16b*) alloc((size_t)M_*H_*2);
  u16b*  FEAT  = (u16b*) alloc((size_t)M_*H_*2);
  u16b*  XP    = (u16b*) alloc((size_t)M_*1024*2);
  u16b*  XC    = (u16b*) alloc((size_t)M_*512*2);
  u16b*  U     = (u16b*) alloc((size_t)M_*64*2);
  u16b*  HS    = (u16b*) alloc((size_t)M_*64*2);
  float* TMP   = (float*)alloc((size_t)M_*256*4);
  u16b*  PAT   = (u16b*) alloc((size_t)M_*256*2);
  float* CARRY = (float*)alloc(8192*4);
  float* HIN   = (float*)alloc(8192*4);
  u16b*  WIN   = (u16b*) alloc((size_t)L_*1024*256*2);
  u16b*  WBP   = (u16b*) alloc((size_t)L_*64*512*2);
  u16b*  WCP   = (u16b*) alloc((size_t)L_*512*64*2);
  u16b*  WOUT  = (u16b*) alloc((size_t)L_*256*512*2);
  u16b*  WFUS  = (u16b*) alloc((size_t)256*512*2);
  u16b*  WHEAD = (u16b*) alloc((size_t)4096*256*2);
  u16b*  WPAT  = (u16b*) alloc((size_t)4*3*64*256*2);

  auto cvt = [&](const float* s, u16b* d, int n){
    cvt_f2bf<<<(n+255)/256, 256, 0, stream>>>(s, d, n);
  };
  cvt(in_w,   WIN,   L_*1024*256);
  cvt(Bp_w,   WBP,   L_*64*512);
  cvt(Cp_w,   WCP,   L_*512*64);
  cvt(out_w,  WOUT,  L_*256*512);
  cvt(fus_w,  WFUS,  256*512);
  cvt(head_w, WHEAD, 4096*256);
  repack_pat<<<(4*64*256*3 + 255)/256, 256, 0, stream>>>(pat_w, WPAT);

  embed_kernel<<<M_, 256, 0, stream>>>(tok, ctx, temb, pemb, poff, EMB, FEAT);

  auto gblocks = [](int N){ return (M_/256) * (N/64); };
  auto shbytes = [](int K){ return (size_t)64 * (K + 8) * 2; };

  for (int i = 0; i < L_; i++){
    // in-proj: XP = FEAT @ in_w[i]^T   (N=1024, K=256)
    gemm_wmma<EPI_BF><<<gblocks(1024), 256, shbytes(256), stream>>>(
        FEAT, nullptr, WIN + (size_t)i*1024*256, 1024, 256, 256,
        XP, nullptr, nullptr, nullptr, nullptr, nullptr, nullptr);
    // depthwise conv + silu -> XC
    dwconv<<<(M_*512)/256, 256, 0, stream>>>(
        XP, conv_w + (size_t)i*512*4, conv_b + (size_t)i*512, XC);
    // u-proj: U = XC @ Bp_w[i]^T   (N=64, K=512)
    gemm_wmma<EPI_BF><<<gblocks(64), 256, shbytes(512), stream>>>(
        XC, nullptr, WBP + (size_t)i*64*512, 64, 512, 512,
        U, nullptr, nullptr, nullptr, nullptr, nullptr, nullptr);
    // parallel scan -> HS
    scan_p1<<<32, 256, 0, stream>>>(U, CARRY);
    scan_p2<<<1, 256, 0, stream>>>(CARRY, HIN);
    scan_p3<<<32, 256, 0, stream>>>(U, HIN, HS);
    // y-proj + D*xc + silu gate -> XC (in place)  (N=512, K=64)
    gemm_wmma<EPI_GATE><<<gblocks(512), 256, shbytes(64), stream>>>(
        HS, nullptr, WCP + (size_t)i*512*64, 512, 64, 64,
        XC, nullptr, nullptr, Dp + (size_t)i*512, XP + INNER_, XC, nullptr);
    // out-proj + residual -> TMP (fp32)  (N=256, K=512)
    gemm_wmma<EPI_RESID><<<gblocks(256), 256, shbytes(512), stream>>>(
        XC, nullptr, WOUT + (size_t)i*256*512, 256, 512, 512,
        nullptr, TMP, nullptr, nullptr, nullptr, nullptr, FEAT);
    // layernorm -> FEAT
    ln_kernel<<<M_, 256, 0, stream>>>(TMP, FEAT, ln_g + (size_t)i*256, ln_b + (size_t)i*256);
  }

  // dilated pattern convs -> PAT
  {
    dim3 g(M_/256, 4);
    pat_gemm<<<g, 256, shbytes(256), stream>>>(EMB, WPAT, pat_b, PAT);
  }
  // fuse: relu([FEAT|PAT] @ fus_w^T + fus_b) -> TMP  (N=256, K=512 concat)
  gemm_wmma<EPI_RELU><<<gblocks(256), 256, shbytes(512), stream>>>(
      FEAT, PAT, WFUS, 256, 512, 256,
      nullptr, TMP, fus_b, nullptr, nullptr, nullptr, nullptr);
  // layernorm -> FUSED (reuse EMB buffer)
  ln_kernel<<<M_, 256, 0, stream>>>(TMP, EMB, fus_g, fus_bb);
  // head: logits[b,c,t,v] = FUSED @ head_w^T + head_b  (N=4096, K=256)
  gemm_wmma<EPI_HEAD><<<gblocks(4096), 256, shbytes(256), stream>>>(
      EMB, nullptr, WHEAD, 4096, 256, 256,
      nullptr, out, head_b, nullptr, nullptr, nullptr, nullptr);

  (void)in_sizes; (void)n_in; (void)out_size; (void)ws_size;
}
